// Block_53369263620290
// MI455X (gfx1250) — matrix operand order repaired and checked by reading, not compiled, not run on hardware
//
#include <hip/hip_runtime.h>
#include <cstdint>

// ---------------- sizes (fixed by the reference) ----------------
#define BB 8
#define TT 1024
#define DD 1024
#define HH 16
#define HS 64
#define EE 8
#define KK 2
#define DF 4096
#define BT (BB * TT)

typedef __attribute__((ext_vector_type(16))) __bf16 v16bf;
typedef __attribute__((ext_vector_type(8)))  float  v8f;
typedef __attribute__((ext_vector_type(4)))  unsigned int u32x4;
typedef __attribute__((ext_vector_type(8)))  int i32x8;
typedef __attribute__((ext_vector_type(4)))  int i32x4;

#if defined(__has_builtin)
#if __has_builtin(__builtin_amdgcn_tensor_load_to_lds) && \
    __has_builtin(__builtin_amdgcn_s_wait_tensorcnt)
#define USE_TDM 1
#endif
#endif
#ifndef USE_TDM
#define USE_TDM 0
#endif

__device__ inline v8f zero8() {
  v8f v;
#pragma unroll
  for (int i = 0; i < 8; ++i) v[i] = 0.f;
  return v;
}

__device__ inline v8f wmma_bf16(v16bf a, v16bf b, v8f c) {
  return __builtin_amdgcn_wmma_f32_16x16x32_bf16(false, a, false, b, (short)0, c,
                                                 false, false);
}

// A fragment 16x32 (MxK), row-major source, p00 -> element (m0,k0).
// lane L<16: M=L, K in {0..7,16..23}; lane L>=16: M=L-16, K in {8..15,24..31}.
// Per lane: two contiguous 16B runs -> 2x b128 loads (alignment asserted).
__device__ inline v16bf load_frag_a(const __bf16* p00, int stride, int lane) {
  int m  = lane & 15;
  int g8 = (lane >> 4) * 8;
  const __bf16* p =
      (const __bf16*)__builtin_assume_aligned(p00 + (size_t)m * stride + g8, 16);
  v16bf a;
#pragma unroll
  for (int j = 0; j < 8; ++j) a[j] = p[j];
#pragma unroll
  for (int j = 0; j < 8; ++j) a[8 + j] = p[16 + j];
  return a;
}

// B fragment 32x16 (KxN) whose SOURCE is stored transposed: S[n][k] row-major.
// lane L: N = L&15, K in {0..7,16..23} (L<16) or {8..15,24..31} (L>=16) (2x b128).
__device__ inline v16bf load_frag_bt(const __bf16* p00, int stride, int lane) {
  int n   = lane & 15;
  int g8  = (lane >> 4) * 8;
  const __bf16* p =
      (const __bf16*)__builtin_assume_aligned(p00 + (size_t)n * stride + g8, 16);
  v16bf b;
#pragma unroll
  for (int j = 0; j < 8; ++j) b[j] = p[j];
#pragma unroll
  for (int j = 0; j < 8; ++j) b[8 + j] = p[16 + j];
  return b;
}

__device__ inline float rsum16(float v) {
#pragma unroll
  for (int m = 8; m >= 1; m >>= 1) v += __shfl_xor(v, m, 16);
  return v;
}
__device__ inline float rmax16(float v) {
#pragma unroll
  for (int m = 8; m >= 1; m >>= 1) v = fmaxf(v, __shfl_xor(v, m, 16));
  return v;
}
__device__ inline float rsum32(float v) {
#pragma unroll
  for (int m = 16; m >= 1; m >>= 1) v += __shfl_xor(v, m, 32);
  return v;
}

union bfpack16 { uint4 v; __bf16 h[8]; };

// ---------------- f32 -> bf16 conversion (plain) ----------------
__global__ __launch_bounds__(256) void cvt_bf16_kernel(const float* __restrict__ in,
                                                       __bf16* __restrict__ out,
                                                       size_t n) {
  size_t i = (size_t)blockIdx.x * blockDim.x + threadIdx.x;
  size_t step = (size_t)gridDim.x * blockDim.x;
  for (; i < n; i += step) out[i] = (__bf16)in[i];
}

// ------- f32 -> bf16 transposing conversion: in [R][C] -> out [C][R] --------
// grid: (C/32, R/32, batch), block (32, 8)
__global__ __launch_bounds__(256) void cvt_t_kernel(const float* __restrict__ in,
                                                    __bf16* __restrict__ out,
                                                    int R, int C) {
  __shared__ __bf16 tile[32][33];
  size_t base = (size_t)blockIdx.z * R * C;
  const float* inb = in + base;
  __bf16* outb = out + base;
  int bx = blockIdx.x * 32;  // C offset
  int by = blockIdx.y * 32;  // R offset
  int tx = threadIdx.x, ty = threadIdx.y;
#pragma unroll
  for (int u = 0; u < 4; ++u) {
    int r = by + ty + u * 8;
    tile[ty + u * 8][tx] = (__bf16)inb[(size_t)r * C + bx + tx];
  }
  __syncthreads();
#pragma unroll
  for (int u = 0; u < 4; ++u) {
    int c = bx + ty + u * 8;
    outb[(size_t)c * R + by + tx] = tile[tx][ty + u * 8];
  }
}

// ---------------- LayerNorm (row = 1024) -> bf16 ----------------
__global__ __launch_bounds__(256) void ln_kernel(const float* __restrict__ x,
                                                 const float* __restrict__ gamma,
                                                 const float* __restrict__ beta,
                                                 __bf16* __restrict__ out) {
  const int D = DD;
  int row = blockIdx.x, tid = threadIdx.x, lane = tid & 31, wave = tid >> 5;
  __shared__ float red[8];
  const float* xr = x + (size_t)row * D;
  float v[4];
  float s = 0.f;
#pragma unroll
  for (int i = 0; i < 4; ++i) { v[i] = xr[tid + 256 * i]; s += v[i]; }
  s = rsum32(s);
  if (lane == 0) red[wave] = s;
  __syncthreads();
  float tot = 0.f;
#pragma unroll
  for (int i = 0; i < 8; ++i) tot += red[i];
  float mean = tot * (1.f / D);
  __syncthreads();
  float vs = 0.f;
#pragma unroll
  for (int i = 0; i < 4; ++i) { float d = v[i] - mean; vs += d * d; }
  vs = rsum32(vs);
  if (lane == 0) red[wave] = vs;
  __syncthreads();
  float vtot = 0.f;
#pragma unroll
  for (int i = 0; i < 8; ++i) vtot += red[i];
  float inv = rsqrtf(vtot * (1.f / D) + 1e-5f);
  __bf16* orow = out + (size_t)row * D;
#pragma unroll
  for (int i = 0; i < 4; ++i) {
    int c = tid + 256 * i;
    orow[c] = (__bf16)((v[i] - mean) * inv * gamma[c] + beta[c]);
  }
}

// ---------------- generic bf16 GEMM: C = A[M,K] @ B[K,N] ----------------
// 128x64 tile, BK=32, 8 waves in 4x2, each wave a 32x32 block (2x2 WMMA frags).
// B tile staged TRANSPOSED in LDS so fragments are contiguous b128 loads.
#define GBM 128
#define GBN 64
#define GBK 32
template <bool BIAS, bool RESID, bool OUTF, bool OUTB>
__global__ __launch_bounds__(256) void gemm_bf16_kernel(
    const __bf16* __restrict__ A, int lda, const __bf16* __restrict__ B, int ldb,
    long long bz, const float* __restrict__ bias, const float* __restrict__ resid,
    float* __restrict__ Cf, __bf16* __restrict__ Cb, int ldc, int czoff, int K) {
  __shared__ __bf16 As[GBM][GBK + 8];   // stride 40 elems = 80B
  __shared__ __bf16 BsT[GBN][GBK + 8];  // transposed: [n][k]
  int z = blockIdx.z;
  const __bf16* Bz = B + (long long)z * bz;
  int m0 = blockIdx.x * GBM;
  int n0 = blockIdx.y * GBN;
  int tid = threadIdx.x, lane = tid & 31, wave = tid >> 5;
  int wm = wave >> 1, wn = wave & 1;
  int g = lane >> 4, cl = lane & 15;

  v8f acc[2][2];
#pragma unroll
  for (int i = 0; i < 2; ++i)
#pragma unroll
    for (int j = 0; j < 2; ++j) acc[i][j] = zero8();

  for (int kk = 0; kk < K; kk += GBK) {
    __syncthreads();
    {  // A tile 128x32 (row-major)
      int r = tid >> 1, c = (tid & 1) * 16;
      const __bf16* src = A + (size_t)(m0 + r) * lda + kk + c;
      *(uint4*)&As[r][c]     = *(const uint4*)src;
      *(uint4*)&As[r][c + 8] = *(const uint4*)(src + 8);
    }
    {  // B tile 32x64, written transposed
      int r = tid >> 3, c = (tid & 7) * 8;  // k=r, n=c..c+7
      bfpack16 u;
      u.v = *(const uint4*)(Bz + (size_t)(kk + r) * ldb + n0 + c);
#pragma unroll
      for (int j = 0; j < 8; ++j) BsT[c + j][r] = u.h[j];
    }
    __syncthreads();
    v16bf a0 = load_frag_a(&As[wm * 32][0], GBK + 8, lane);
    v16bf a1 = load_frag_a(&As[wm * 32 + 16][0], GBK + 8, lane);
    v16bf b0 = load_frag_bt(&BsT[wn * 32][0], GBK + 8, lane);
    v16bf b1 = load_frag_bt(&BsT[wn * 32 + 16][0], GBK + 8, lane);
    acc[0][0] = wmma_bf16(a0, b0, acc[0][0]);
    acc[0][1] = wmma_bf16(a0, b1, acc[0][1]);
    acc[1][0] = wmma_bf16(a1, b0, acc[1][0]);
    acc[1][1] = wmma_bf16(a1, b1, acc[1][1]);
  }
#pragma unroll
  for (int mi = 0; mi < 2; ++mi)
#pragma unroll
    for (int ni = 0; ni < 2; ++ni) {
      int row0 = m0 + wm * 32 + mi * 16 + 8 * g;
      int col  = n0 + z * czoff + wn * 32 + ni * 16 + cl;
      float bval = BIAS ? bias[col] : 0.f;
#pragma unroll
      for (int i = 0; i < 8; ++i) {
        size_t idx = (size_t)(row0 + i) * ldc + col;
        float v = acc[mi][ni][i] + bval;
        if (RESID) v += resid[idx];
        if (OUTF) Cf[idx] = v;
        if (OUTB) Cb[idx] = (__bf16)v;
      }
    }
}

// ---------------- causal flash attention (per (b,h), 128 q rows / WG) -------
__global__ __launch_bounds__(256) void attn_kernel(const __bf16* __restrict__ Q,
                                                   const __bf16* __restrict__ Km,
                                                   const __bf16* __restrict__ V,
                                                   __bf16* __restrict__ O) {
  __shared__ __bf16 Ks[64][64 + 8];    // row-major [key][hs]
  __shared__ __bf16 VsT[64][64 + 8];   // transposed [hs][key]
  __shared__ __bf16 Ps[8][16][64 + 8];
  int bh = blockIdx.y;
  int b = bh / HH, h = bh % HH;
  int tid = threadIdx.x, lane = tid & 31, wave = tid >> 5;
  int g = lane >> 4, cl = lane & 15;
  size_t rowbase = (size_t)b * TT;
  int colbase = h * HS;
  int qr0 = blockIdx.x * 128 + wave * 16;

  v16bf qa[2];
#pragma unroll
  for (int ks = 0; ks < 2; ++ks)
    qa[ks] = load_frag_a(Q + (rowbase + qr0) * DD + colbase + ks * 32, DD, lane);

  float mrow[8], lrow[8];
  v8f o[4];
#pragma unroll
  for (int i = 0; i < 8; ++i) { mrow[i] = -INFINITY; lrow[i] = 0.f; }
#pragma unroll
  for (int nt = 0; nt < 4; ++nt) o[nt] = zero8();

  int jb_end = (blockIdx.x * 128 + 127) / 64;
  for (int jb = 0; jb <= jb_end; ++jb) {
    __syncthreads();
    {  // stage K row-major, V transposed
      int r = tid >> 2, c = (tid & 3) * 16;
      const __bf16* ksrc = Km + (rowbase + jb * 64 + r) * DD + colbase + c;
      const __bf16* vsrc = V  + (rowbase + jb * 64 + r) * DD + colbase + c;
      *(uint4*)&Ks[r][c]     = *(const uint4*)ksrc;
      *(uint4*)&Ks[r][c + 8] = *(const uint4*)(ksrc + 8);
      bfpack16 u0, u1;
      u0.v = *(const uint4*)vsrc;
      u1.v = *(const uint4*)(vsrc + 8);
#pragma unroll
      for (int j = 0; j < 8; ++j) VsT[c + j][r]     = u0.h[j];
#pragma unroll
      for (int j = 0; j < 8; ++j) VsT[c + 8 + j][r] = u1.h[j];
    }
    __syncthreads();

    v8f s[4];
#pragma unroll
    for (int nt = 0; nt < 4; ++nt) {
      s[nt] = zero8();
#pragma unroll
      for (int ks = 0; ks < 2; ++ks)
        s[nt] = wmma_bf16(qa[ks], load_frag_bt(&Ks[nt * 16][ks * 32], 64 + 8, lane),
                          s[nt]);
    }
    const float sc = 0.03125f;  // D^-0.5 (reference scales by n_embed^-0.5)
    float mcur[8];
#pragma unroll
    for (int i = 0; i < 8; ++i) mcur[i] = -INFINITY;
#pragma unroll
    for (int nt = 0; nt < 4; ++nt)
#pragma unroll
      for (int i = 0; i < 8; ++i) {
        int key = jb * 64 + nt * 16 + cl;
        int q   = qr0 + i + 8 * g;
        float val = s[nt][i] * sc;
        if (key > q) val = -INFINITY;
        s[nt][i] = val;
        mcur[i]  = fmaxf(mcur[i], val);
      }
    float alpha[8], psum[8];
#pragma unroll
    for (int i = 0; i < 8; ++i) {
      mcur[i] = rmax16(mcur[i]);
      float mn = fmaxf(mrow[i], mcur[i]);
      alpha[i] = __expf(mrow[i] - mn);
      mrow[i]  = mn;
      psum[i]  = 0.f;
    }
#pragma unroll
    for (int nt = 0; nt < 4; ++nt)
#pragma unroll
      for (int i = 0; i < 8; ++i) {
        float p = __expf(s[nt][i] - mrow[i]);
        s[nt][i] = p;
        psum[i] += p;
      }
#pragma unroll
    for (int i = 0; i < 8; ++i) {
      psum[i] = rsum16(psum[i]);
      lrow[i] = lrow[i] * alpha[i] + psum[i];
    }
#pragma unroll
    for (int nt = 0; nt < 4; ++nt)
#pragma unroll
      for (int i = 0; i < 8; ++i) o[nt][i] *= alpha[i];

    // restage P into A-fragment layout (per-wave LDS, same-wave RAW)
#pragma unroll
    for (int nt = 0; nt < 4; ++nt)
#pragma unroll
      for (int i = 0; i < 8; ++i)
        Ps[wave][i + 8 * g][nt * 16 + cl] = (__bf16)s[nt][i];
    v16bf pa0 = load_frag_a(&Ps[wave][0][0], 64 + 8, lane);
    v16bf pa1 = load_frag_a(&Ps[wave][0][32], 64 + 8, lane);
#pragma unroll
    for (int nt = 0; nt < 4; ++nt) {
      o[nt] = wmma_bf16(pa0, load_frag_bt(&VsT[nt * 16][0], 64 + 8, lane), o[nt]);
      o[nt] = wmma_bf16(pa1, load_frag_bt(&VsT[nt * 16][32], 64 + 8, lane), o[nt]);
    }
  }
#pragma unroll
  for (int nt = 0; nt < 4; ++nt)
#pragma unroll
    for (int i = 0; i < 8; ++i) {
      int q = qr0 + i + 8 * g;
      int hs = nt * 16 + cl;
      O[(rowbase + q) * DD + colbase + hs] = (__bf16)(o[nt][i] / lrow[i]);
    }
}

// ---------------- router: logits, noisy top-2, dispatch lists ----------------
__global__ void zero_counts_kernel(int* counts) {
  if (threadIdx.x < EE) counts[threadIdx.x] = 0;
}

__global__ __launch_bounds__(256) void router_kernel(
    const __bf16* __restrict__ h2, const float* __restrict__ Wr,
    const float* __restrict__ br, const float* __restrict__ Wn,
    const float* __restrict__ bn, const float* __restrict__ noise,
    int* __restrict__ counts, int* __restrict__ disp_tok,
    float* __restrict__ disp_gate) {
  int tid = threadIdx.x, lane = tid & 31, wave = tid >> 5;
  int row = blockIdx.x * 8 + wave;
  const __bf16* hr = h2 + (size_t)row * DD;
  float accr[EE], accn[EE];
#pragma unroll
  for (int e = 0; e < EE; ++e) { accr[e] = 0.f; accn[e] = 0.f; }
  for (int d = lane; d < DD; d += 32) {
    float hv = (float)hr[d];
    const float* wr = Wr + (size_t)d * EE;
    const float* wn = Wn + (size_t)d * EE;
#pragma unroll
    for (int e = 0; e < EE; ++e) { accr[e] += hv * wr[e]; accn[e] += hv * wn[e]; }
  }
#pragma unroll
  for (int e = 0; e < EE; ++e) { accr[e] = rsum32(accr[e]); accn[e] = rsum32(accn[e]); }
  if (lane == 0) {
    float nl[EE];
#pragma unroll
    for (int e = 0; e < EE; ++e) {
      float si = accn[e] + bn[e];
      float sp = (si > 20.f) ? si : log1pf(__expf(si));
      nl[e] = accr[e] + br[e] + noise[(size_t)row * EE + e] * sp;
    }
    int i0 = 0;
#pragma unroll
    for (int e = 1; e < EE; ++e) if (nl[e] > nl[i0]) i0 = e;
    int i1 = (i0 == 0) ? 1 : 0;
#pragma unroll
    for (int e = 0; e < EE; ++e) if (e != i0 && nl[e] > nl[i1]) i1 = e;
    float e1 = __expf(nl[i1] - nl[i0]);
    float z  = 1.f + e1;
    float g0 = 1.f / z, g1 = e1 / z;
    int p0 = atomicAdd(&counts[i0], 1);
    disp_tok[i0 * BT + p0]  = row;
    disp_gate[i0 * BT + p0] = g0;
    int p1 = atomicAdd(&counts[i1], 1);
    disp_tok[i1 * BT + p1]  = row;
    disp_gate[i1 * BT + p1] = g1;
  }
}

// ---------------- grouped MoE FFN: 16 gathered tokens x 1 expert / WG -------
// W1t: per expert [DF][D] (= W1^T), W2t: per expert [D][DF] (= W2^T) so all
// B fragments are read as two contiguous 16B runs (2x global b128 per lane).
// h2 rows gathered into LDS by the Tensor Data Mover (gather mode, 16-bit
// indices), padded 16B per 1KB -> row stride 1040 elems, cols >=512 shift +8.
#define H2RS 1040
__global__ __launch_bounds__(256) void moe_kernel(
    const __bf16* __restrict__ h2, const __bf16* __restrict__ W1t,
    const __bf16* __restrict__ W2t, const float* __restrict__ b1,
    const float* __restrict__ b2, const int* __restrict__ counts,
    const int* __restrict__ disp_tok, const float* __restrict__ disp_gate,
    float* __restrict__ out) {
  int e = blockIdx.y;
  int cnt = counts[e];
  int t0 = blockIdx.x * 16;
  if (t0 >= cnt) return;

  __shared__ __attribute__((aligned(16))) __bf16 h2s_raw[16 * H2RS];
  __shared__ __bf16 acts[16][128 + 8];
  __shared__ int    stok[16];
  __shared__ float  sgate[16];

  int tid = threadIdx.x, lane = tid & 31, wave = tid >> 5;
  int g = lane >> 4, cl = lane & 15;

  if (tid < 16) {  // lanes 0-15 of wave 0
    int idx = t0 + tid;
    stok[tid]  = (idx < cnt) ? disp_tok[e * BT + idx] : disp_tok[e * BT + t0];
    sgate[tid] = (idx < cnt) ? disp_gate[e * BT + idx] : 0.f;
  }
#if USE_TDM
  if (wave == 0) {
    // Build D# (ISA 8.3-8.7) and issue a gather-mode TDM load of the 16 token
    // rows (16x1024 bf16) into LDS; completion via TENSORcnt.
    uint32_t lds_off = (uint32_t)(uintptr_t)(void*)h2s_raw;
    uint64_t ga = (uint64_t)(uintptr_t)(const void*)h2;
    u32x4 g0;
    g0[0] = 1u | (1u << 31);  // count=1, gather_mode=1, 16-bit indices
    g0[1] = lds_off;          // lds_addr
    g0[2] = (uint32_t)ga;     // global_addr[31:0]
    g0[3] = (uint32_t)((ga >> 32) & 0x1FFFFFFu) | (2u << 30);  // [56:32], type=2
    i32x8 g1;
    g1[0] = (int)((1u << 16) |          // data_size = 2 bytes
                  (1u << 20) |          // pad_enable
                  (7u << 22) |          // pad_interval: every 256 dwords (1KB)
                  (3u << 25));          // pad_amount: 4 dwords (16B)
    g1[1] = (int)(1024u << 16);         // tensor_dim0 = 1024 (lo16)
    g1[2] = (int)((uint32_t)BT << 16);  // dim0 hi16=0 | tensor_dim1 lo16
    g1[3] = (int)(1024u << 16);         // dim1 hi16=0 | tile_dim0 = 1024
    g1[4] = 16;                         // tile_dim1 = #valid indices
    g1[5] = 1024;                       // tensor_dim0_stride = 1024
    g1[6] = 0;
    g1[7] = 0;
    i32x4 g2, g3;
#pragma unroll
    for (int j = 0; j < 4; ++j)
      g2[j] = (stok[2 * j] & 0xFFFF) | (stok[2 * j + 1] << 16);
#pragma unroll
    for (int j = 0; j < 4; ++j)
      g3[j] = (stok[8 + 2 * j] & 0xFFFF) | (stok[9 + 2 * j] << 16);
    i32x8 g4;  // extra descriptor group (6-arg toolchain form): unused -> zero
#pragma unroll
    for (int j = 0; j < 8; ++j) g4[j] = 0;
    __builtin_amdgcn_tensor_load_to_lds(g0, g1, g2, g3, g4, 0);
    __builtin_amdgcn_s_wait_tensorcnt((short)0);
  }
  __syncthreads();
#else
  __syncthreads();
  {  // manual gather into the same padded layout
    int r = tid >> 4, c0 = (tid & 15) * 64;
    const __bf16* src = h2 + (size_t)stok[r] * DD + c0;
    __bf16* dst = h2s_raw + (size_t)r * H2RS + c0 + ((c0 >= 512) ? 8 : 0);
#pragma unroll
    for (int u = 0; u < 8; ++u)
      *(uint4*)(dst + u * 8) = *(const uint4*)(src + u * 8);
  }
#endif
  const __bf16* W1te = W1t + (size_t)e * DF * DD;  // [DF][D]
  const __bf16* W2te = W2t + (size_t)e * DD * DF;  // [D][DF]
  const float*  b1e = b1 + (size_t)e * DF;
  const float*  b2e = b2 + (size_t)e * DD;

  v8f oacc[8];
#pragma unroll
  for (int nt = 0; nt < 8; ++nt) oacc[nt] = zero8();
  int ncol0 = wave * 128;

  for (int ch = 0; ch < DF; ch += 128) {
    __syncthreads();  // protect acts reuse (and first h2s read)
    if (ch + 128 < DF) {  // prefetch next chunk's weights
      __builtin_prefetch(W1te + (size_t)(ch + 128 + wave * 16) * DD, 0, 0);
      __builtin_prefetch(W2te + (size_t)ncol0 * DF + ch + 128, 0, 0);
    }
    // GEMM1: this wave computes act columns [ch + wave*16, +16)
    v8f c1 = zero8();
    int w1col = ch + wave * 16;
    for (int k0 = 0; k0 < DD; k0 += 32) {
      int kadj = k0 + ((k0 >= 512) ? 8 : 0);  // mid-row LDS pad
      v16bf a = load_frag_a(h2s_raw + kadj, H2RS, lane);
      v16bf b = load_frag_bt(W1te + (size_t)w1col * DD + k0, DD, lane);
      c1 = wmma_bf16(a, b, c1);
    }
#pragma unroll
    for (int i = 0; i < 8; ++i) {
      float v = c1[i] + b1e[w1col + cl];
      v = v > 0.f ? v : 0.f;
      acts[i + 8 * g][wave * 16 + cl] = (__bf16)v;
    }
    __syncthreads();
    // GEMM2: out[16, wave*128 .. +128) += act[16,128] @ W2[ch..ch+128, :]
#pragma unroll
    for (int ks = 0; ks < 4; ++ks) {
      v16bf pa = load_frag_a(&acts[0][ks * 32], 128 + 8, lane);
#pragma unroll
      for (int nt = 0; nt < 8; ++nt) {
        v16bf b = load_frag_bt(W2te + (size_t)(ncol0 + nt * 16) * DF + ch + ks * 32,
                               DF, lane);
        oacc[nt] = wmma_bf16(pa, b, oacc[nt]);
      }
    }
  }
#pragma unroll
  for (int nt = 0; nt < 8; ++nt)
#pragma unroll
    for (int i = 0; i < 8; ++i) {
      int r = i + 8 * g;
      int col = ncol0 + nt * 16 + cl;
      float v = (oacc[nt][i] + b2e[col]) * sgate[r];
      atomicAdd(&out[(size_t)stok[r] * DD + col], v);
    }
}

// ---------------- host orchestration ----------------
extern "C" void kernel_launch(void* const* d_in, const int* in_sizes, int n_in,
                              void* d_out, int out_size, void* d_ws, size_t ws_size,
                              hipStream_t stream) {
  const float* x     = (const float*)d_in[0];
  const float* noise = (const float*)d_in[1];
  const float* Wq    = (const float*)d_in[2];
  const float* Wk    = (const float*)d_in[3];
  const float* Wv    = (const float*)d_in[4];
  const float* Wproj = (const float*)d_in[5];
  const float* bproj = (const float*)d_in[6];
  const float* ln1_g = (const float*)d_in[7];
  const float* ln1_b = (const float*)d_in[8];
  const float* ln2_g = (const float*)d_in[9];
  const float* ln2_b = (const float*)d_in[10];
  const float* Wr    = (const float*)d_in[11];
  const float* br    = (const float*)d_in[12];
  const float* Wn    = (const float*)d_in[13];
  const float* bn    = (const float*)d_in[14];
  const float* W1    = (const float*)d_in[15];
  const float* b1    = (const float*)d_in[16];
  const float* W2    = (const float*)d_in[17];
  const float* b2    = (const float*)d_in[18];
  float* out = (float*)d_out;

  char* ws = (char*)d_ws;
  size_t off = 0;
  auto alloc = [&](size_t bytes) -> char* {
    char* p = ws + off;
    off += (bytes + 255) & ~(size_t)255;
    return p;
  };
  const size_t act_elems = (size_t)BT * DD;
  __bf16* hb      = (__bf16*)alloc(act_elems * 2);
  __bf16* qb      = (__bf16*)alloc(act_elems * 2);
  __bf16* kb      = (__bf16*)alloc(act_elems * 2);
  __bf16* vb      = (__bf16*)alloc(act_elems * 2);
  __bf16* ob      = (__bf16*)alloc(act_elems * 2);
  __bf16* h2b     = (__bf16*)alloc(act_elems * 2);
  __bf16* wq_b    = (__bf16*)alloc((size_t)HH * DD * HS * 2);
  __bf16* wk_b    = (__bf16*)alloc((size_t)HH * DD * HS * 2);
  __bf16* wv_b    = (__bf16*)alloc((size_t)HH * DD * HS * 2);
  __bf16* wproj_b = (__bf16*)alloc((size_t)DD * DD * 2);
  __bf16* w1t_b   = (__bf16*)alloc((size_t)EE * DD * DF * 2);
  __bf16* w2t_b   = (__bf16*)alloc((size_t)EE * DF * DD * 2);
  int*    counts    = (int*)alloc(EE * sizeof(int));
  int*    disp_tok  = (int*)alloc((size_t)EE * BT * sizeof(int));
  float*  disp_gate = (float*)alloc((size_t)EE * BT * sizeof(float));
  (void)ws_size; (void)in_sizes; (void)n_in; (void)out_size;

  auto cvt = [&](const float* src, __bf16* dst, size_t n) {
    int blocks = (int)((n + 1023) / 1024);
    cvt_bf16_kernel<<<blocks, 256, 0, stream>>>(src, dst, n);
  };
  cvt(Wq, wq_b, (size_t)HH * DD * HS);
  cvt(Wk, wk_b, (size_t)HH * DD * HS);
  cvt(Wv, wv_b, (size_t)HH * DD * HS);
  cvt(Wproj, wproj_b, (size_t)DD * DD);
  {  // W1 [E][D][DF] -> W1t [E][DF][D]
    dim3 grid(DF / 32, DD / 32, EE), block(32, 8);
    cvt_t_kernel<<<grid, block, 0, stream>>>(W1, w1t_b, DD, DF);
  }
  {  // W2 [E][DF][D] -> W2t [E][D][DF]
    dim3 grid(DD / 32, DF / 32, EE), block(32, 8);
    cvt_t_kernel<<<grid, block, 0, stream>>>(W2, w2t_b, DF, DD);
  }

  // LN1: x -> h (bf16)
  ln_kernel<<<BT, 256, 0, stream>>>(x, ln1_g, ln1_b, hb);

  // Q/K/V per-head GEMMs: [BT,1024] x [1024,64] via grid.z = head
  {
    dim3 grid(BT / GBM, 1, HH);
    gemm_bf16_kernel<false, false, false, true><<<grid, 256, 0, stream>>>(
        hb, DD, wq_b, HS, (long long)DD * HS, nullptr, nullptr, nullptr, qb, DD,
        HS, DD);
    gemm_bf16_kernel<false, false, false, true><<<grid, 256, 0, stream>>>(
        hb, DD, wk_b, HS, (long long)DD * HS, nullptr, nullptr, nullptr, kb, DD,
        HS, DD);
    gemm_bf16_kernel<false, false, false, true><<<grid, 256, 0, stream>>>(
        hb, DD, wv_b, HS, (long long)DD * HS, nullptr, nullptr, nullptr, vb, DD,
        HS, DD);
  }

  // causal attention, heads concatenated into ob[BT, D]
  {
    dim3 grid(TT / 128, BB * HH);
    attn_kernel<<<grid, 256, 0, stream>>>(qb, kb, vb, ob);
  }

  // x2 = x + ob @ Wproj + bproj  -> d_out (f32)
  {
    dim3 grid(BT / GBM, DD / GBN, 1);
    gemm_bf16_kernel<true, true, true, false><<<grid, 256, 0, stream>>>(
        ob, DD, wproj_b, DD, 0, bproj, x, out, nullptr, DD, 0, DD);
  }

  // LN2: x2 -> h2 (bf16)
  ln_kernel<<<BT, 256, 0, stream>>>(out, ln2_g, ln2_b, h2b);

  // router + dispatch lists
  zero_counts_kernel<<<1, 32, 0, stream>>>(counts);
  router_kernel<<<BT / 8, 256, 0, stream>>>(h2b, Wr, br, Wn, bn, noise, counts,
                                            disp_tok, disp_gate);

  // grouped MoE FFN, gate-weighted atomic accumulation into d_out
  {
    dim3 grid(BT / 16, EE);
    moe_kernel<<<grid, 256, 0, stream>>>(h2b, w1t_b, w2t_b, b1, b2, counts,
                                         disp_tok, disp_gate, out);
  }
}
